// ScaledDotProductAttention_34016140984994
// MI455X (gfx1250) — matrix operand order repaired and checked by reading, not compiled, not run on hardware
//
#include <hip/hip_runtime.h>
#include <stdint.h>

// CDNA5 (gfx1250) flash-style attention forward.
// Path A (preferred, needs 16MiB workspace): prep kernel converts K->bf16 and
//   V->bf16 transposed per 32-key tile; main kernel streams tiles into LDS with
//   the Tensor Data Mover (tensor_load_to_lds, TENSORcnt), double-buffered, and
//   runs the math on v_wmma_f32_16x16x32_bf16 with fp32 online softmax.
// Path B (fallback): all-in-one kernel (in-loop f32->bf16 conversion).
// Faithful to the reference's multiplicative tril mask: masked scores are 0.0
// (NOT -inf), so every key tile participates in softmax and in P*V.

typedef __attribute__((ext_vector_type(16))) __bf16 v16bf;
typedef __attribute__((ext_vector_type(8)))  float  v8f;
typedef __attribute__((ext_vector_type(4)))  unsigned int v4u;
typedef __attribute__((ext_vector_type(8)))  int v8i;
typedef __attribute__((ext_vector_type(4)))  int v4i;

#define HD    128            // head dim
#define SEQ   2048           // sequence length
#define QTILE 128            // query rows per block (8 waves x 16 rows)
#define KT    32             // keys per main-loop iteration
#define NJ    (SEQ / KT)     // 64 key tiles
#define WAVES 8
#define TILE_ELEMS (KT * HD) // 4096 bf16 = 8KB per tile

#if defined(__HIP_DEVICE_COMPILE__) && \
    __has_builtin(__builtin_amdgcn_tensor_load_to_lds) && \
    __has_builtin(__builtin_amdgcn_s_wait_tensorcnt)
#define HAVE_TDM 1
#else
#define HAVE_TDM 0
#endif

__device__ __forceinline__ uint16_t f2bf(float f) {
    union { float f; uint32_t u; } x; x.f = f;
    uint32_t u = x.u;
    uint32_t r = u + 0x7FFFu + ((u >> 16) & 1u);   // round-to-nearest-even
    return (uint16_t)(r >> 16);
}

union FragB {                 // one 16x32 bf16 A-frag or 32x16 bf16 B-frag
    uint4 q[2];               // 32 bytes
    v16bf v;
};

#if HAVE_TDM
// Issue one TDM load of a contiguous 8KB (4096 x 2B) block into LDS.
// D# per CDNA5 ISA: 1-D tensor, count=1, type=2 (image),
// data_size=1 (2 bytes), tile_dim0 = tensor_dim0 = 4096, no pad/iterate/mcast.
__device__ __forceinline__ void tdm_load_tile8k(uint32_t lds_off, const uint16_t* gsrc) {
    const uint64_t ga = (uint64_t)(uintptr_t)gsrc;
    v4u g0;
    g0.x = 1u;                                                  // count=1 (user mode)
    g0.y = lds_off;                                             // lds_addr (bytes)
    g0.z = (uint32_t)ga;                                        // global_addr[31:0]
    g0.w = (uint32_t)((ga >> 32) & 0x01FFFFFFu) | (2u << 30);   // addr[56:32] | type=2
    v8i g1;
    g1[0] = 0x00010000;              // data_size=1 (2B); no multicast/pad/iterate
    g1[1] = (int)(4096u << 16);      // tensor_dim0 = 4096 (low 16 bits at bit 48)
    g1[2] = 0x00010000;              // tensor_dim0 hi = 0 | tensor_dim1 = 1
    g1[3] = (int)(4096u << 16);      // tensor_dim1 hi = 0 | tile_dim0 = 4096
    g1[4] = 1;                       // tile_dim1 = 1, tile_dim2 = 0
    g1[5] = 4096;                    // tensor_dim0_stride = 4096
    g1[6] = 0;
    g1[7] = 0;
    v4i g2 = (v4i)0;                 // 2-D use: groups 2/3 zero
    v4i g3 = (v4i)0;
#if __clang_major__ >= 23
    v8i g4 = (v8i)0;
    __builtin_amdgcn_tensor_load_to_lds(g0, g1, g2, g3, g4, 0);
#else
    __builtin_amdgcn_tensor_load_to_lds(g0, g1, g2, g3, 0);
#endif
}
#endif

// ---- Q fragments: A-matrix layout (16x32 bf16) from fp32 global, done once ----
// lane: m = ln, half = lg; element e: k = 32f + 16*lg + e
__device__ __forceinline__ void load_q(FragB* qf, const float* qptr, int lg) {
    #pragma unroll
    for (int f = 0; f < 4; ++f) {
        #pragma unroll
        for (int half = 0; half < 2; ++half) {
            const int k0 = 32 * f + 16 * lg + 8 * half;
            uint32_t w[4];
            #pragma unroll
            for (int p = 0; p < 4; ++p) {
                float2 v2 = *(const float2*)(qptr + k0 + 2 * p);
                w[p] = (uint32_t)f2bf(v2.x) | ((uint32_t)f2bf(v2.y) << 16);
            }
            qf[f].q[half] = make_uint4(w[0], w[1], w[2], w[3]);
        }
    }
}

// ---- one 32-key tile: S = QK^T (8 WMMA), masked online softmax, O += P*V (8 WMMA) ----
// Fragment loads are software-pipelined (2-deep) so ds_load_b128 latency hides
// under the WMMA pipe (wait dscnt<=2 instead of 0).
__device__ __forceinline__ void attn_tile(
    v8f* acc, float* rmax, float* rsum, const FragB* qf,
    const uint16_t* skt,   // K tile  [key(32)][d(128)]  bf16
    const uint16_t* svt,   // V tile  [d(128)][key(32)]  bf16 (transposed)
    uint16_t* pw,          // per-wave P staging [m(16)][k(32)]
    int kbase, int qrow0, int ln, int lg)
{
    const float scale = 0.08838834764831845f;   // 1/sqrt(128)

    // ---- S = Q K^T : two 16x16 C tiles, alternating accumulators ----
    v8f sc[2];
    #pragma unroll
    for (int tc = 0; tc < 2; ++tc)
        #pragma unroll
        for (int i = 0; i < 8; ++i) sc[tc][i] = 0.0f;

    FragB kf[2];
    {   // prefetch (tc=0, f=0): B-frag n=ln, element e -> d = 16*lg + e
        const uint4* s = (const uint4*)(skt + ln * HD + 16 * lg);
        kf[0].q[0] = s[0];
        kf[0].q[1] = s[1];
    }
    #pragma unroll
    for (int i = 0; i < 8; ++i) {
        const int tc = i & 1;
        const int f  = i >> 1;
        if (i + 1 < 8) {   // prefetch next fragment into the other buffer
            const int ntc = (i + 1) & 1;
            const int nf  = (i + 1) >> 1;
            const uint4* s = (const uint4*)(skt + (16 * ntc + ln) * HD + 32 * nf + 16 * lg);
            kf[(i + 1) & 1].q[0] = s[0];
            kf[(i + 1) & 1].q[1] = s[1];
        }
        sc[tc] = __builtin_amdgcn_wmma_f32_16x16x32_bf16(
                     false, qf[f].v, false, kf[i & 1].v, (short)0, sc[tc], false, false);
    }

    // ---- scale + multiplicative tril mask (masked score := 0.0f) ----
    float tred[8];
    #pragma unroll
    for (int r = 0; r < 8; ++r) {
        const int qg = qrow0 + r + 8 * lg;
        float mx = -INFINITY;
        #pragma unroll
        for (int tc = 0; tc < 2; ++tc) {
            const int kg = kbase + 16 * tc + ln;
            float s = sc[tc][r] * scale;
            if (kg > qg) s = 0.0f;
            sc[tc][r] = s;
            mx = fmaxf(mx, s);
        }
        tred[r] = mx;
    }
    #pragma unroll
    for (int sh = 1; sh < 16; sh <<= 1) {
        #pragma unroll
        for (int r = 0; r < 8; ++r)
            tred[r] = fmaxf(tred[r], __shfl_xor(tred[r], sh, 32));
    }

    float alpha[8];
    #pragma unroll
    for (int r = 0; r < 8; ++r) {
        const float mnew = fmaxf(rmax[r], tred[r]);
        alpha[r] = __expf(rmax[r] - mnew);     // exp(-inf)=0 on first tile
        rmax[r]  = mnew;
        float psum = 0.0f;
        #pragma unroll
        for (int tc = 0; tc < 2; ++tc) {
            const float p = __expf(sc[tc][r] - mnew);
            sc[tc][r] = p;
            psum += p;
        }
        tred[r] = psum;
    }
    #pragma unroll
    for (int sh = 1; sh < 16; sh <<= 1) {
        #pragma unroll
        for (int r = 0; r < 8; ++r)
            tred[r] += __shfl_xor(tred[r], sh, 32);
    }
    #pragma unroll
    for (int r = 0; r < 8; ++r) rsum[r] = rsum[r] * alpha[r] + tred[r];
    #pragma unroll
    for (int c = 0; c < 8; ++c)
        #pragma unroll
        for (int r = 0; r < 8; ++r) acc[c][r] *= alpha[r];

    // ---- O += P (16x32) * V (32x128) ----
    // prefetch first V-frag before the P staging stores so its latency overlaps
    FragB vfr[2];
    {
        const uint4* s = (const uint4*)(svt + ln * KT + 16 * lg);   // c = 0
        vfr[0].q[0] = s[0];
        vfr[0].q[1] = s[1];
    }

    // P: C-layout -> A-layout via per-wave LDS bounce (same-wave LDS is in-order)
    #pragma unroll
    for (int tc = 0; tc < 2; ++tc)
        #pragma unroll
        for (int r = 0; r < 8; ++r)
            pw[(r + 8 * lg) * KT + 16 * tc + ln] = f2bf(sc[tc][r]);

    FragB pf;
    pf.q[0] = *(const uint4*)(pw + ln * KT + 16 * lg);       // e<8 : k = 16*lg + e
    pf.q[1] = *(const uint4*)(pw + ln * KT + 16 * lg + 8);   // e>=8: k = 16*lg + e

    #pragma unroll
    for (int c = 0; c < 8; ++c) {
        if (c + 1 < 8) {   // prefetch next V-frag: n = d = 16(c+1)+ln, e -> key = 16*lg+e
            const uint4* s = (const uint4*)(svt + (16 * (c + 1) + ln) * KT + 16 * lg);
            vfr[(c + 1) & 1].q[0] = s[0];
            vfr[(c + 1) & 1].q[1] = s[1];
        }
        acc[c] = __builtin_amdgcn_wmma_f32_16x16x32_bf16(
                     false, pf.v, false, vfr[c & 1].v, (short)0, acc[c], false, false);
    }
}

__device__ __forceinline__ void store_out(
    const v8f* acc, const float* rsum, float* O, int b, int qrow0, int ln, int lg)
{
    #pragma unroll
    for (int r = 0; r < 8; ++r) {
        const int qg = qrow0 + r + 8 * lg;
        const float inv = 1.0f / rsum[r];
        float* op = O + ((size_t)b * SEQ + qg) * HD;
        #pragma unroll
        for (int c = 0; c < 8; ++c)
            op[16 * c + ln] = acc[c][r] * inv;
    }
}

// ===================== prep: K,V fp32 -> bf16 (V tile-transposed) =====================
__global__ __launch_bounds__(256) void prep_bf16(
    const float* __restrict__ K, const float* __restrict__ V,
    uint16_t* __restrict__ Kb, uint16_t* __restrict__ Vt)
{
    const size_t idx = (size_t)blockIdx.x * 256 + threadIdx.x;  // over B*SEQ*HD
    Kb[idx] = f2bf(K[idx]);
    const int    d    = (int)(idx & (HD - 1));
    const size_t slin = idx >> 7;                 // b*SEQ + s
    const int    key  = (int)(slin & (KT - 1));
    const size_t tile = slin >> 5;                // b*NJ + j
    Vt[(tile * HD + d) * KT + key] = f2bf(V[idx]);
}

// ===================== main kernel, TDM-fed double-buffered tiles =====================
__global__ __launch_bounds__(256) void fa_fwd_tdm(
    const float* __restrict__ Q, const uint16_t* __restrict__ Kb,
    const uint16_t* __restrict__ Vt, float* __restrict__ O)
{
    __shared__ __align__(16) uint16_t kbuf[2][TILE_ELEMS];   // [key][d]
    __shared__ __align__(16) uint16_t vbuf[2][TILE_ELEMS];   // [d][key]
    __shared__ __align__(16) uint16_t spt[WAVES][16 * KT];

    const int b   = blockIdx.y;
    const int qb  = blockIdx.x * QTILE;
    const int t   = threadIdx.x;
    const int wid = t >> 5;
    const int l   = t & 31;
    const int ln  = l & 15;
    const int lg  = l >> 4;
    const int qrow0 = qb + wid * 16;

    FragB qf[4];
    load_q(qf, Q + ((size_t)b * SEQ + qrow0 + ln) * HD, lg);

    v8f acc[8];
    #pragma unroll
    for (int c = 0; c < 8; ++c)
        #pragma unroll
        for (int i = 0; i < 8; ++i) acc[c][i] = 0.0f;
    float rmax[8], rsum[8];
    #pragma unroll
    for (int r = 0; r < 8; ++r) { rmax[r] = -INFINITY; rsum[r] = 0.0f; }

    const uint16_t* ktile0 = Kb + (size_t)b * SEQ * HD;        // tile j at +j*TILE_ELEMS
    const uint16_t* vtile0 = Vt + (size_t)b * NJ * TILE_ELEMS;

#if HAVE_TDM
    if (wid == 0) {   // wave 0 drives the Tensor Data Mover
        tdm_load_tile8k((uint32_t)(uintptr_t)&kbuf[0][0], ktile0);
        tdm_load_tile8k((uint32_t)(uintptr_t)&vbuf[0][0], vtile0);
    }
    for (int j = 0; j < NJ; ++j) {
        const int cur = j & 1;
        if (wid == 0) {
            if (j + 1 < NJ) {   // prefetch next tile pair, then wait for current pair
                tdm_load_tile8k((uint32_t)(uintptr_t)&kbuf[cur ^ 1][0],
                                ktile0 + (size_t)(j + 1) * TILE_ELEMS);
                tdm_load_tile8k((uint32_t)(uintptr_t)&vbuf[cur ^ 1][0],
                                vtile0 + (size_t)(j + 1) * TILE_ELEMS);
                __builtin_amdgcn_s_wait_tensorcnt(2);
            } else {
                __builtin_amdgcn_s_wait_tensorcnt(0);
            }
        }
        __syncthreads();   // tile j visible to all waves
        attn_tile(acc, rmax, rsum, qf, kbuf[cur], vbuf[cur], spt[wid],
                  j * KT, qrow0, ln, lg);
        __syncthreads();   // all waves done reading buf[cur] before its next overwrite
    }
#else
    for (int j = 0; j < NJ; ++j) {
        __syncthreads();
        const uint4* gk = (const uint4*)(ktile0 + (size_t)j * TILE_ELEMS);
        const uint4* gv = (const uint4*)(vtile0 + (size_t)j * TILE_ELEMS);
        uint4* lk = (uint4*)&kbuf[0][0];
        uint4* lv = (uint4*)&vbuf[0][0];
        #pragma unroll
        for (int i = 0; i < 2; ++i) {
            lk[i * 256 + t] = gk[i * 256 + t];
            lv[i * 256 + t] = gv[i * 256 + t];
        }
        __syncthreads();
        attn_tile(acc, rmax, rsum, qf, kbuf[0], vbuf[0], spt[wid],
                  j * KT, qrow0, ln, lg);
    }
#endif
    store_out(acc, rsum, O, b, qrow0, ln, lg);
}

// ===================== fallback: all-in-one (no workspace) =====================
__global__ __launch_bounds__(256) void fa_fwd_fused(
    const float* __restrict__ Q, const float* __restrict__ K,
    const float* __restrict__ V, float* __restrict__ O)
{
    __shared__ __align__(16) uint16_t skt[TILE_ELEMS];
    __shared__ __align__(16) uint16_t svt[TILE_ELEMS];
    __shared__ __align__(16) uint16_t spt[WAVES][16 * KT];

    const int b   = blockIdx.y;
    const int qb  = blockIdx.x * QTILE;
    const int t   = threadIdx.x;
    const int wid = t >> 5;
    const int l   = t & 31;
    const int ln  = l & 15;
    const int lg  = l >> 4;
    const int qrow0 = qb + wid * 16;

    FragB qf[4];
    load_q(qf, Q + ((size_t)b * SEQ + qrow0 + ln) * HD, lg);

    v8f acc[8];
    #pragma unroll
    for (int c = 0; c < 8; ++c)
        #pragma unroll
        for (int i = 0; i < 8; ++i) acc[c][i] = 0.0f;
    float rmax[8], rsum[8];
    #pragma unroll
    for (int r = 0; r < 8; ++r) { rmax[r] = -INFINITY; rsum[r] = 0.0f; }

    for (int j = 0; j < NJ; ++j) {
        const int kbase = j * KT;
        __syncthreads();
        #pragma unroll
        for (int i = 0; i < TILE_ELEMS / 256; ++i) {
            const int idx = i * 256 + t;
            const int key = idx >> 7;
            const int d   = idx & (HD - 1);
            const size_t g = ((size_t)b * SEQ + kbase + key) * HD + d;
            skt[key * HD + d] = f2bf(K[g]);
            svt[d * KT + key] = f2bf(V[g]);
        }
        __syncthreads();
        attn_tile(acc, rmax, rsum, qf, skt, svt, spt[wid], kbase, qrow0, ln, lg);
    }
    store_out(acc, rsum, O, b, qrow0, ln, lg);
}

extern "C" void kernel_launch(void* const* d_in, const int* in_sizes, int n_in,
                              void* d_out, int out_size, void* d_ws, size_t ws_size,
                              hipStream_t stream) {
    const float* Q = (const float*)d_in[0];
    const float* K = (const float*)d_in[1];
    const float* V = (const float*)d_in[2];
    float* O = (float*)d_out;
    const int B = in_sizes[0] / (SEQ * HD);
    const size_t nelem = (size_t)B * SEQ * HD;
    const size_t need  = 2 * nelem * sizeof(uint16_t);

    dim3 grid(SEQ / QTILE, B);
    dim3 block(256);

    if (ws_size >= need) {
        uint16_t* Kb = (uint16_t*)d_ws;
        uint16_t* Vt = Kb + nelem;
        prep_bf16<<<dim3((unsigned)(nelem / 256)), block, 0, stream>>>(K, V, Kb, Vt);
        fa_fwd_tdm<<<grid, block, 0, stream>>>(Q, Kb, Vt, O);
    } else {
        fa_fwd_fused<<<grid, block, 0, stream>>>(Q, K, V, O);
    }
}
